// CausalSelfAttention_53197464928556
// MI455X (gfx1250) — matrix operand order repaired and checked by reading, not compiled, not run on hardware
//
#include <hip/hip_runtime.h>

typedef __attribute__((ext_vector_type(16))) __bf16   v16bf;
typedef __attribute__((ext_vector_type(8)))  float    v8f;
typedef __attribute__((ext_vector_type(4)))  unsigned u32x4;
typedef __attribute__((ext_vector_type(4)))  float    f32x4;
typedef __attribute__((ext_vector_type(8)))  int      i32x8;
typedef __attribute__((ext_vector_type(4)))  int      i32x4;

#if defined(__has_builtin)
#  if __has_builtin(__builtin_amdgcn_tensor_load_to_lds)
#    define HAVE_TDM 1
#  endif
#endif
#ifndef HAVE_TDM
#  define HAVE_TDM 0
#endif

union Frag {
    v16bf v;
    u32x4 q[2];
    unsigned short s[16];
};

__device__ __forceinline__ unsigned short f2bf(float f) {
    unsigned u = __float_as_uint(f);
    unsigned r = u + 0x7FFFu + ((u >> 16) & 1u);
    return (unsigned short)(r >> 16);
}

#define WMMA_BF16(A, B, Cc) \
    __builtin_amdgcn_wmma_f32_16x16x32_bf16(false, (A), false, (B), (short)0, (Cc), false, false)

#if HAVE_TDM
// Issue one 2D TDM tile load: global (row-major, 2-byte elems, row stride
// `stride0` elems) -> LDS contiguous.  D# layout per CDNA5 ISA ch.8.
// This toolchain exposes the 6-arg builtin: (g0, g1, g2, g3, extra, cpol).
__device__ __forceinline__ void tdm_load_2d(unsigned lds_off, const void* gptr,
                                            unsigned d0, unsigned d1, unsigned stride0)
{
    unsigned long long ga = (unsigned long long)(uintptr_t)gptr;
    u32x4 g0;
    g0[0] = 1u;                                   // count=1 (valid), user mode
    g0[1] = lds_off;                              // lds_addr
    g0[2] = (unsigned)(ga & 0xFFFFFFFFu);         // global_addr[31:0]
    g0[3] = (unsigned)((ga >> 32) & 0x01FFFFFFu)  // global_addr[56:32]
            | 0x80000000u;                        // type=2 ("image")
    i32x8 g1;
    g1[0] = (int)(1u << 16);                      // data_size=1 (2 bytes)
    g1[1] = (int)((d0 & 0xFFFFu) << 16);          // tensor_dim0[15:0]
    g1[2] = (int)(((d0 >> 16) & 0xFFFFu) | ((d1 & 0xFFFFu) << 16));   // td0 hi | td1 lo
    g1[3] = (int)(((d1 >> 16) & 0xFFFFu) | ((d0 & 0xFFFFu) << 16));   // td1 hi | tile_dim0
    g1[4] = (int)(d1 & 0xFFFFu);                  // tile_dim1 (tile_dim2=0)
    g1[5] = (int)stride0;                         // tensor_dim0_stride[31:0]
    g1[6] = 0;                                    // stride0 hi | stride1 lo
    g1[7] = 0;
    i32x4 z4 = {0, 0, 0, 0};                      // groups 2/3 unused (2D tile)
    i32x8 z8 = {0, 0, 0, 0, 0, 0, 0, 0};
    __builtin_amdgcn_tensor_load_to_lds(g0, g1, z4, z4, z8, 0);
}
__device__ __forceinline__ void tdm_wait0() {
#if defined(__has_builtin) && __has_builtin(__builtin_amdgcn_s_wait_tensorcnt)
    __builtin_amdgcn_s_wait_tensorcnt(0);
#else
    asm volatile("s_wait_tensorcnt 0x0" ::: "memory");
#endif
}
#endif

// -------------------------------------------------------------------------
// Kernel 1: QKV GEMM.  x[8192,1024] @ w_qkv[1024,3072] + b_qkv
// -> q_ws,k_ws [BH,T,D] bf16 ; v_ws [BH,D,T] bf16 (transposed for PV B-frags)
// Double-buffered LDS, one barrier per K-step.
// -------------------------------------------------------------------------
__global__ __launch_bounds__(256, 2) void qkv_gemm_kernel(
    const float* __restrict__ x, const float* __restrict__ w,
    const float* __restrict__ bias,
    unsigned short* __restrict__ q_ws, unsigned short* __restrict__ k_ws,
    unsigned short* __restrict__ v_ws)
{
    __shared__ alignas(16) unsigned short lA[2][128][32];   // x tile  [m][k]
    __shared__ alignas(16) unsigned short lB[2][128][32];   // W^T tile [n][k]

    const int tid = threadIdx.x;
    const int wave = tid >> 5, lane = tid & 31;
    const int l16 = lane & 15, hl = lane >> 4;
    const int m0 = blockIdx.y * 128, n0 = blockIdx.x * 128;
    const int wm = wave >> 1, wn = wave & 1;   // 4x2 wave grid: 32 rows x 64 cols
    const int Cdim = 1024, N3 = 3072;

    auto load_tiles = [&](int k0, int buf) {
#pragma unroll
        for (int i = 0; i < 4; i++) {               // A: 128x32 f32 -> bf16
            int c = i * 256 + tid;
            int row = c >> 3, col4 = (c & 7) * 4;
            f32x4 f = *(const f32x4*)(x + (size_t)(m0 + row) * Cdim + k0 + col4);
            unsigned p0 = f2bf(f.x) | ((unsigned)f2bf(f.y) << 16);
            unsigned p1 = f2bf(f.z) | ((unsigned)f2bf(f.w) << 16);
            *(unsigned*)&lA[buf][row][col4]     = p0;
            *(unsigned*)&lA[buf][row][col4 + 2] = p1;
        }
#pragma unroll
        for (int i = 0; i < 4; i++) {               // W^T: lB[n][k] = w[k0+k][n0+n]
            int c = i * 256 + tid;
            int k = c >> 5, n4 = (c & 31) * 4;
            f32x4 f = *(const f32x4*)(w + (size_t)(k0 + k) * N3 + n0 + n4);
            lB[buf][n4 + 0][k] = f2bf(f.x);
            lB[buf][n4 + 1][k] = f2bf(f.y);
            lB[buf][n4 + 2][k] = f2bf(f.z);
            lB[buf][n4 + 3][k] = f2bf(f.w);
        }
    };

    v8f zero = {};
    v8f acc[2][4];
#pragma unroll
    for (int i = 0; i < 2; i++)
#pragma unroll
        for (int j = 0; j < 4; j++) acc[i][j] = zero;

    load_tiles(0, 0);
    __syncthreads();

    for (int kt = 0; kt < 32; kt++) {
        const int cur = kt & 1;
        if (kt + 1 < 32) load_tiles((kt + 1) * 32, cur ^ 1);   // prefetch next tile

        Frag a[2], b[4];
#pragma unroll
        for (int mt = 0; mt < 2; mt++) {
            const unsigned short* p = &lA[cur][wm * 32 + mt * 16 + l16][0];
            a[mt].q[0] = *(const u32x4*)(p + 8 * hl);        // K {0..7}/{8..15}
            a[mt].q[1] = *(const u32x4*)(p + 16 + 8 * hl);   // K {16..23}/{24..31}
        }
#pragma unroll
        for (int nt = 0; nt < 4; nt++) {
            const unsigned short* p = &lB[cur][wn * 64 + nt * 16 + l16][0];
            b[nt].q[0] = *(const u32x4*)(p + 8 * hl);        // K {0..7}/{8..15}
            b[nt].q[1] = *(const u32x4*)(p + 16 + 8 * hl);   // K {16..23}/{24..31}
        }
#pragma unroll
        for (int mt = 0; mt < 2; mt++)
#pragma unroll
            for (int nt = 0; nt < 4; nt++)
                acc[mt][nt] = WMMA_BF16(a[mt].v, b[nt].v, acc[mt][nt]);
        __syncthreads();
    }

    // Epilogue: bias + scatter to head-major q/k and d-major v
    const int T_ = 2048, H_ = 16, D_ = 64;
#pragma unroll
    for (int mt = 0; mt < 2; mt++) {
#pragma unroll
        for (int nt = 0; nt < 4; nt++) {
            int col = n0 + wn * 64 + nt * 16 + l16;    // [0,3072)
            int which = col >> 10;
            int cc = col & 1023;
            int hh = cc >> 6, d = cc & 63;
            float bv = bias[col];
#pragma unroll
            for (int r = 0; r < 8; r++) {
                int row = m0 + wm * 32 + mt * 16 + r + 8 * hl;   // token [0,8192)
                int bidx = row >> 11, tt = row & 2047;
                unsigned short bf = f2bf(acc[mt][nt][r] + bv);
                size_t bh = (size_t)bidx * H_ + hh;
                if (which == 0)      q_ws[(bh * T_ + tt) * D_ + d] = bf;
                else if (which == 1) k_ws[(bh * T_ + tt) * D_ + d] = bf;
                else                 v_ws[(bh * D_ + d) * T_ + tt] = bf;
            }
        }
    }
}

// -------------------------------------------------------------------------
// Kernel 2: causal flash attention. 1 block = 1 head x 128 query rows.
// Each of 8 waves owns 16 query rows.  K/V tiles are DMA'd by the Tensor
// Data Mover (wave 0 issues, TENSORcnt + barrier fences), double-buffered
// one key-block ahead.  o_ws is [B,T,C] bf16.
// -------------------------------------------------------------------------
__global__ __launch_bounds__(256, 2) void attn_kernel(
    const unsigned short* __restrict__ q_ws, const unsigned short* __restrict__ k_ws,
    const unsigned short* __restrict__ v_ws, unsigned short* __restrict__ o_ws)
{
    __shared__ alignas(16) unsigned short lQ[128][64];
    __shared__ alignas(16) unsigned short lK[2][32][64];    // [key][d]
    __shared__ alignas(16) unsigned short lV[2][64][32];    // [d][key] (from v_ws^T)
    __shared__ alignas(16) unsigned short lP[8][16][32];    // per-wave P bounce

    const int tid = threadIdx.x;
    const int wave = tid >> 5, lane = tid & 31;
    const int l16 = lane & 15, hl = lane >> 4;
    const int qt = blockIdx.x, bh = blockIdx.y;
    const int T_ = 2048, D_ = 64, C_ = 1024;
    const int q0 = qt * 128;
    const int nkb = (q0 + 128) >> 5;      // causal limit: key blocks of 32

#if HAVE_TDM
    auto issue_kv = [&](int kb, int buf) {
        int key0 = kb * 32;
        tdm_load_2d((unsigned)(uintptr_t)&lK[buf][0][0],
                    k_ws + (size_t)bh * T_ * D_ + (size_t)key0 * D_,
                    /*d0=*/64, /*d1=*/32, /*stride0=*/64);
        tdm_load_2d((unsigned)(uintptr_t)&lV[buf][0][0],
                    v_ws + (size_t)bh * D_ * T_ + key0,
                    /*d0=*/32, /*d1=*/64, /*stride0=*/2048);
    };
    if (wave == 0) issue_kv(0, 0);        // kick off first K/V tile early
#endif

    // Q tile -> LDS (all threads)
    const unsigned short* qbase = q_ws + (size_t)bh * T_ * D_ + (size_t)q0 * D_;
#pragma unroll
    for (int i = 0; i < 4; i++) {
        int c = i * 256 + tid;
        int row = c >> 3, col8 = (c & 7) * 8;
        *(u32x4*)&lQ[row][col8] = *(const u32x4*)(qbase + row * D_ + col8);
    }
    __syncthreads();

    // Per-wave Q A-fragments (held in registers for the whole kernel)
    Frag qa[2];
#pragma unroll
    for (int dh = 0; dh < 2; dh++) {
        const unsigned short* p = &lQ[wave * 16 + l16][dh * 32];
        qa[dh].q[0] = *(const u32x4*)(p + 8 * hl);
        qa[dh].q[1] = *(const u32x4*)(p + 16 + 8 * hl);
    }

    v8f zero = {};
    v8f O[4];
#pragma unroll
    for (int i = 0; i < 4; i++) O[i] = zero;
    float mrow[8], lrow[8];
#pragma unroll
    for (int r = 0; r < 8; r++) { mrow[r] = -3.0e38f; lrow[r] = 0.f; }

    for (int kb = 0; kb < nkb; ++kb) {
        const int key0 = kb * 32;
        const int cur = kb & 1;
#if HAVE_TDM
        if (wave == 0) tdm_wait0();       // this wave's DMA for buf[cur] done
        __syncthreads();                  // visible to all; prev reads of buf^1 done
        if (kb + 1 < nkb && wave == 0) issue_kv(kb + 1, cur ^ 1);
#else
        {   // cooperative K (32x64) and V^T (64x32) loads
            int row = tid >> 3, col8 = (tid & 7) * 8;
            *(u32x4*)&lK[cur][row][col8] =
                *(const u32x4*)(k_ws + (size_t)bh * T_ * D_ + (size_t)(key0 + row) * D_ + col8);
            int d = tid >> 2, c8 = (tid & 3) * 8;
            *(u32x4*)&lV[cur][d][c8] =
                *(const u32x4*)(v_ws + (size_t)bh * D_ * T_ + (size_t)d * T_ + key0 + c8);
        }
        __syncthreads();
#endif

        // S = Q @ K^T : two 16x16 tiles, K-dim = d (64) -> 2 WMMAs each
        v8f S[2];
#pragma unroll
        for (int st = 0; st < 2; st++) {
            v8f s = zero;
#pragma unroll
            for (int dh = 0; dh < 2; dh++) {
                Frag kf;
                const unsigned short* p = &lK[cur][st * 16 + l16][dh * 32];
                kf.q[0] = *(const u32x4*)(p + 8 * hl);
                kf.q[1] = *(const u32x4*)(p + 16 + 8 * hl);
                s = WMMA_BF16(qa[dh].v, kf.v, s);
            }
            S[st] = s;
        }

        // scale + causal mask + row max
        float rmax[8];
#pragma unroll
        for (int r = 0; r < 8; r++) rmax[r] = -3.0e38f;
#pragma unroll
        for (int st = 0; st < 2; st++) {
            int key = key0 + st * 16 + l16;
#pragma unroll
            for (int r = 0; r < 8; r++) {
                int qi = q0 + wave * 16 + r + 8 * hl;
                float v = S[st][r] * 0.125f;            // 1/sqrt(64)
                v = (key <= qi) ? v : -3.0e38f;
                S[st][r] = v;
                rmax[r] = fmaxf(rmax[r], v);
            }
        }
#pragma unroll
        for (int r = 0; r < 8; r++) {                   // reduce within 16-lane group
            float v = rmax[r];
            v = fmaxf(v, __shfl_xor(v, 1));
            v = fmaxf(v, __shfl_xor(v, 2));
            v = fmaxf(v, __shfl_xor(v, 4));
            v = fmaxf(v, __shfl_xor(v, 8));
            rmax[r] = v;
        }
        float alpha[8];
#pragma unroll
        for (int r = 0; r < 8; r++) {
            float mnew = fmaxf(mrow[r], rmax[r]);
            alpha[r] = __expf(mrow[r] - mnew);
            mrow[r] = mnew;
        }
        float rsum[8];
#pragma unroll
        for (int r = 0; r < 8; r++) rsum[r] = 0.f;
#pragma unroll
        for (int st = 0; st < 2; st++)
#pragma unroll
            for (int r = 0; r < 8; r++) {
                float p = __expf(S[st][r] - mrow[r]);
                S[st][r] = p;
                rsum[r] += p;
            }
#pragma unroll
        for (int r = 0; r < 8; r++) {
            float v = rsum[r];
            v += __shfl_xor(v, 1);
            v += __shfl_xor(v, 2);
            v += __shfl_xor(v, 4);
            v += __shfl_xor(v, 8);
            lrow[r] = lrow[r] * alpha[r] + v;
        }
#pragma unroll
        for (int dt = 0; dt < 4; dt++)
#pragma unroll
            for (int r = 0; r < 8; r++) O[dt][r] *= alpha[r];

        // P: C-layout -> A-layout via per-wave LDS bounce (wave-local, in-order)
#pragma unroll
        for (int st = 0; st < 2; st++)
#pragma unroll
            for (int r = 0; r < 8; r++)
                lP[wave][r + 8 * hl][st * 16 + l16] = f2bf(S[st][r]);
        Frag pa;
        {
            const unsigned short* p = &lP[wave][l16][0];
            pa.q[0] = *(const u32x4*)(p + 8 * hl);
            pa.q[1] = *(const u32x4*)(p + 16 + 8 * hl);
        }
        // O += P(16x32) @ V(32x64) : 4 WMMAs (one per 16-wide d tile)
#pragma unroll
        for (int dt = 0; dt < 4; dt++) {
            Frag vf;
            const unsigned short* p = &lV[cur][dt * 16 + l16][0];
            vf.q[0] = *(const u32x4*)(p + 8 * hl);
            vf.q[1] = *(const u32x4*)(p + 16 + 8 * hl);
            O[dt] = WMMA_BF16(pa.v, vf.v, O[dt]);
        }
#if !HAVE_TDM
        __syncthreads();
#endif
    }

    // normalize + store to [B,T,C] bf16
    int b = bh >> 4, hh = bh & 15;
#pragma unroll
    for (int r = 0; r < 8; r++) {
        float inv = 1.0f / lrow[r];
        int qi = q0 + wave * 16 + r + 8 * hl;
        size_t rowbase = ((size_t)b * T_ + qi) * C_ + hh * D_;
#pragma unroll
        for (int dt = 0; dt < 4; dt++)
            o_ws[rowbase + dt * 16 + l16] = f2bf(O[dt][r] * inv);
    }
}

// -------------------------------------------------------------------------
// Kernel 3: output projection. attn[8192,1024] bf16 @ w_proj[1024,1024] + b
// Double-buffered LDS, one barrier per K-step.
// -------------------------------------------------------------------------
__global__ __launch_bounds__(256, 2) void proj_gemm_kernel(
    const unsigned short* __restrict__ a_bf, const float* __restrict__ w,
    const float* __restrict__ bias, float* __restrict__ out)
{
    __shared__ alignas(16) unsigned short lA[2][128][32];
    __shared__ alignas(16) unsigned short lB[2][128][32];

    const int tid = threadIdx.x;
    const int wave = tid >> 5, lane = tid & 31;
    const int l16 = lane & 15, hl = lane >> 4;
    const int m0 = blockIdx.y * 128, n0 = blockIdx.x * 128;
    const int wm = wave >> 1, wn = wave & 1;
    const int Cdim = 1024;

    auto load_tiles = [&](int k0, int buf) {
#pragma unroll
        for (int i = 0; i < 2; i++) {          // A tile: bf16 straight copy
            int c = i * 256 + tid;
            int row = c >> 2, col8 = (c & 3) * 8;
            *(u32x4*)&lA[buf][row][col8] =
                *(const u32x4*)(a_bf + (size_t)(m0 + row) * Cdim + k0 + col8);
        }
#pragma unroll
        for (int i = 0; i < 4; i++) {          // W^T tile with f32->bf16
            int c = i * 256 + tid;
            int k = c >> 5, n4 = (c & 31) * 4;
            f32x4 f = *(const f32x4*)(w + (size_t)(k0 + k) * Cdim + n0 + n4);
            lB[buf][n4 + 0][k] = f2bf(f.x);
            lB[buf][n4 + 1][k] = f2bf(f.y);
            lB[buf][n4 + 2][k] = f2bf(f.z);
            lB[buf][n4 + 3][k] = f2bf(f.w);
        }
    };

    v8f zero = {};
    v8f acc[2][4];
#pragma unroll
    for (int i = 0; i < 2; i++)
#pragma unroll
        for (int j = 0; j < 4; j++) acc[i][j] = zero;

    load_tiles(0, 0);
    __syncthreads();

    for (int kt = 0; kt < 32; kt++) {
        const int cur = kt & 1;
        if (kt + 1 < 32) load_tiles((kt + 1) * 32, cur ^ 1);

        Frag a[2], b[4];
#pragma unroll
        for (int mt = 0; mt < 2; mt++) {
            const unsigned short* p = &lA[cur][wm * 32 + mt * 16 + l16][0];
            a[mt].q[0] = *(const u32x4*)(p + 8 * hl);
            a[mt].q[1] = *(const u32x4*)(p + 16 + 8 * hl);
        }
#pragma unroll
        for (int nt = 0; nt < 4; nt++) {
            const unsigned short* p = &lB[cur][wn * 64 + nt * 16 + l16][0];
            b[nt].q[0] = *(const u32x4*)(p + 8 * hl);
            b[nt].q[1] = *(const u32x4*)(p + 16 + 8 * hl);
        }
#pragma unroll
        for (int mt = 0; mt < 2; mt++)
#pragma unroll
            for (int nt = 0; nt < 4; nt++)
                acc[mt][nt] = WMMA_BF16(a[mt].v, b[nt].v, acc[mt][nt]);
        __syncthreads();
    }

#pragma unroll
    for (int mt = 0; mt < 2; mt++)
#pragma unroll
        for (int nt = 0; nt < 4; nt++) {
            int col = n0 + wn * 64 + nt * 16 + l16;
            float bv = bias[col];
#pragma unroll
            for (int r = 0; r < 8; r++) {
                int row = m0 + wm * 32 + mt * 16 + r + 8 * hl;
                out[(size_t)row * Cdim + col] = acc[mt][nt][r] + bv;
            }
        }
}

// -------------------------------------------------------------------------
extern "C" void kernel_launch(void* const* d_in, const int* in_sizes, int n_in,
                              void* d_out, int out_size, void* d_ws, size_t ws_size,
                              hipStream_t stream)
{
    const float* x      = (const float*)d_in[0];
    const float* w_qkv  = (const float*)d_in[1];
    const float* b_qkv  = (const float*)d_in[2];
    const float* w_proj = (const float*)d_in[3];
    const float* b_proj = (const float*)d_in[4];
    float* out = (float*)d_out;

    // workspace: 4 bf16 buffers of BH*T*D == B*T*C == 8,388,608 elements each
    unsigned short* ws   = (unsigned short*)d_ws;
    const size_t seg = (size_t)64 * 2048 * 64;
    unsigned short* q_ws = ws;
    unsigned short* k_ws = q_ws + seg;
    unsigned short* v_ws = k_ws + seg;
    unsigned short* o_ws = v_ws + seg;

    dim3 blk(256);
    dim3 g1(3072 / 128, 8192 / 128);   // 24 x 64
    qkv_gemm_kernel<<<g1, blk, 0, stream>>>(x, w_qkv, b_qkv, q_ws, k_ws, v_ws);

    dim3 g2(2048 / 128, 4 * 16);       // 16 q-tiles x 64 heads
    attn_kernel<<<g2, blk, 0, stream>>>(q_ws, k_ws, v_ws, o_ws);

    dim3 g3(1024 / 128, 8192 / 128);   // 8 x 64
    proj_gemm_kernel<<<g3, blk, 0, stream>>>(o_ws, w_proj, b_proj, out);
}
